// CyberFinGAT_84284438217028
// MI455X (gfx1250) — matrix operand order repaired and checked by reading, not compiled, not run on hardware
//
#include <hip/hip_runtime.h>

typedef __attribute__((ext_vector_type(16))) _Float16 v16h;
typedef __attribute__((ext_vector_type(8)))  _Float16 v8h;
typedef __attribute__((ext_vector_type(8)))  float    v8f;

#define DHEAD   32
#define LEAKY   0.2f
#define BN_EPSF 1e-5f

__device__ __forceinline__ void atomic_max_f32(float* addr, float v) {
  // sign-aware float max via integer atomics (works with -inf init)
  if (v >= 0.0f) atomicMax((int*)addr, __float_as_int(v));
  else           atomicMin((unsigned int*)addr, __float_as_uint(v));
}

// ---------------- layer-1 GEMM: [N,8] x [8,HD] (K too small for WMMA) ------
__global__ void k_gemm_in8(const float* __restrict__ x, const float* __restrict__ W,
                           float* __restrict__ h, int N, int HD) {
  int idx = blockIdx.x * blockDim.x + threadIdx.x;
  if (idx >= N * HD) return;
  int n = idx / HD, f = idx - n * HD;
  const float* xr = x + (size_t)n * 8;
  float s = 0.f;
#pragma unroll
  for (int k = 0; k < 8; ++k) s += xr[k] * W[k * HD + f];
  h[idx] = s;
}

// ---------------- W [K,Nc] f32 -> Wt [Nc,K] f16 (transpose + convert) ------
__global__ void k_transpose_f16(const float* __restrict__ W, _Float16* __restrict__ Wt,
                                int K, int Nc) {
  int i = blockIdx.x * blockDim.x + threadIdx.x;
  if (i >= K * Nc) return;
  int c = i / K, k = i - c * K;                // output-major: Wt[c][k]
  Wt[(size_t)c * K + k] = (_Float16)W[(size_t)k * Nc + c];
}

// ---------------- WMMA GEMM: A[M,K]f16 x Wt[Nc,K]f16 -> C[M,Nc]f32 ---------
// One wave per 16x16 output tile; K stepped by 32 via v_wmma_f32_16x16x32_f16.
// Both A and B fragments are contiguous 16B vector loads (B is pre-transposed).
__global__ void k_gemm_wmma(const _Float16* __restrict__ A, const _Float16* __restrict__ Wt,
                            float* __restrict__ C, int M, int Nc, int K, int totalTiles) {
  int wave = threadIdx.x >> 5;
  int lane = threadIdx.x & 31;
  int tile = blockIdx.x * (blockDim.x >> 5) + wave;
  if (tile >= totalTiles) return;          // wave-uniform: EXEC stays all-1s for WMMA
  int tilesN = Nc >> 4;
  int tm = tile / tilesN, tn = tile - tm * tilesN;
  int row = tm * 16 + (lane & 15);
  int rrow = (row < M) ? row : (M - 1);    // clamp (M is a multiple of 16 here anyway)
  int col = tn * 16 + (lane & 15);
  // A 16x32 f16 layout: lanes 0-15 hold K {0..7,16..23}; lanes 16-31 hold K {8..15,24..31}
  int kbA = (lane < 16) ? 0 : 8;
  // B 32x16 f16 layout: lanes 0-15 hold K {0..7,16..23}; lanes 16-31 hold K {8..15,24..31}
  int kbB = (lane < 16) ? 0 : 8;
  const _Float16* apBase = A  + (size_t)rrow * K + kbA;   // +k0 per step
  const _Float16* bpBase = Wt + (size_t)col  * K + kbB;   // +k0 per step (contiguous K!)
  v8f acc = {};
  for (int k0 = 0; k0 < K; k0 += 32) {
    const _Float16* ap = apBase + k0;
    const _Float16* bp = bpBase + k0;
    if (k0 + 32 < K) {                      // prefetch next K-step (global_prefetch_b8)
      __builtin_prefetch(ap + 32, 0, 1);
      __builtin_prefetch(bp + 32, 0, 1);
    }
    v8h a0 = *(const v8h*)(ap);             // K = k0+kbA+0..7
    v8h a1 = *(const v8h*)(ap + 16);        // K = k0+kbA+16..23
    v8h b0 = *(const v8h*)(bp);             // K = k0+kbB+0..7
    v8h b1 = *(const v8h*)(bp + 16);        // K = k0+kbB+16..23
    v16h a, b;
#pragma unroll
    for (int i = 0; i < 8; ++i) {
      a[i] = a0[i]; a[i + 8] = a1[i];
      b[i] = b0[i]; b[i + 8] = b1[i];
    }
    acc = __builtin_amdgcn_wmma_f32_16x16x32_f16(false, a, false, b, (short)0, acc,
                                                 false, false);
  }
  // C/D layout: VGPR r -> row (tm*16 + r + (lane<16?0:8)), col = lane&15
  int mr = tm * 16 + ((lane < 16) ? 0 : 8);
#pragma unroll
  for (int r = 0; r < 8; ++r) {
    int orow = mr + r;
    if (orow < M) C[(size_t)orow * Nc + col] = acc[r];
  }
}

// ---------------- attention logits: al_s/al_d = <h[n,h,:], a[h,:]> ---------
__global__ void k_att_dots(const float* __restrict__ h, const float* __restrict__ a_src,
                           const float* __restrict__ a_dst, float* __restrict__ als,
                           float* __restrict__ ald, int N, int H) {
  int idx = blockIdx.x * blockDim.x + threadIdx.x;
  if (idx >= N * H) return;
  int n = idx / H, hh = idx - n * H;
  const float* hp  = h + (size_t)n * H * DHEAD + hh * DHEAD;
  const float* asp = a_src + hh * DHEAD;
  const float* adp = a_dst + hh * DHEAD;
  float s = 0.f, d = 0.f;
#pragma unroll
  for (int k = 0; k < DHEAD; ++k) { s += hp[k] * asp[k]; d += hp[k] * adp[k]; }
  als[idx] = s;
  ald[idx] = d;
}

// ---------------- init accumulators ----------------------------------------
__global__ void k_init(float* __restrict__ out, float* __restrict__ m,
                       float* __restrict__ den, int N, int HD, int H) {
  int idx = blockIdx.x * blockDim.x + threadIdx.x;
  if (idx < N * HD) out[idx] = 0.f;
  if (idx < N * H) { m[idx] = -__builtin_inff(); den[idx] = 0.f; }
}

// ---------------- edge pass 1: segment max of leaky_relu(logit) ------------
__global__ void k_edge_max(const long long* __restrict__ ei, int E, int N,
                           const float* __restrict__ als, const float* __restrict__ ald,
                           float* __restrict__ m, int H) {
  int e = blockIdx.x * blockDim.x + threadIdx.x;
  if (e >= E + N) return;
  int s, d;
  if (e < E) { s = (int)ei[e]; d = (int)ei[(size_t)E + e]; } else { s = d = e - E; }
  for (int hh = 0; hh < H; ++hh) {
    float v = als[s * H + hh] + ald[d * H + hh];
    v = (v < 0.f) ? LEAKY * v : v;
    atomic_max_f32(&m[d * H + hh], v);
  }
}

// ---------------- edge pass 2: segment sum of exp(e - m[dst]) --------------
__global__ void k_edge_den(const long long* __restrict__ ei, int E, int N,
                           const float* __restrict__ als, const float* __restrict__ ald,
                           const float* __restrict__ m, float* __restrict__ den, int H) {
  int e = blockIdx.x * blockDim.x + threadIdx.x;
  if (e >= E + N) return;
  int s, d;
  if (e < E) { s = (int)ei[e]; d = (int)ei[(size_t)E + e]; } else { s = d = e - E; }
  for (int hh = 0; hh < H; ++hh) {
    float v = als[s * H + hh] + ald[d * H + hh];
    v = (v < 0.f) ? LEAKY * v : v;
    atomicAdd(&den[d * H + hh], expf(v - m[d * H + hh]));
  }
}

// ---------------- edge pass 3: out[dst] += alpha * h[src] (wave per edge) --
__global__ void k_edge_agg(const long long* __restrict__ ei, int E, int N,
                           const float* __restrict__ als, const float* __restrict__ ald,
                           const float* __restrict__ m, const float* __restrict__ den,
                           const float* __restrict__ h, float* __restrict__ out,
                           int H, int HD) {
  int wave = threadIdx.x >> 5, lane = threadIdx.x & 31;
  int e = blockIdx.x * (blockDim.x >> 5) + wave;
  if (e >= E + N) return;
  int s, d;
  if (e < E) { s = (int)ei[e]; d = (int)ei[(size_t)E + e]; } else { s = d = e - E; }
  int fpl = HD >> 5;          // features per lane: 4 (HD=128) or 1 (HD=32)
  int f0 = lane * fpl;
  int hh = f0 / DHEAD;        // lane's features never straddle a head
  float v = als[s * H + hh] + ald[d * H + hh];
  v = (v < 0.f) ? LEAKY * v : v;
  float alpha = expf(v - m[d * H + hh]) / (den[d * H + hh] + 1e-16f);
  const float* hs = h + (size_t)s * HD + f0;
  float* op = out + (size_t)d * HD + f0;
  for (int i = 0; i < fpl; ++i) atomicAdd(&op[i], alpha * hs[i]);
}

// ---------------- bias + BN + ELU, write f16 activations --------------------
__global__ void k_finalize_h16(const float* __restrict__ out, const float* __restrict__ bias,
                               const float* __restrict__ g, const float* __restrict__ be,
                               const float* __restrict__ mean, const float* __restrict__ var,
                               _Float16* __restrict__ xh, int N, int HD) {
  int idx = blockIdx.x * blockDim.x + threadIdx.x;
  if (idx >= N * HD) return;
  int f = idx % HD;
  float y = out[idx] + bias[f];
  y = (y - mean[f]) * rsqrtf(var[f] + BN_EPSF) * g[f] + be[f];
  y = (y > 0.f) ? y : expm1f(y);
  xh[idx] = (_Float16)y;
}

// ---------------- bias + BN + ELU, write f32 (layer 3) ----------------------
__global__ void k_finalize_f32(const float* __restrict__ out, const float* __restrict__ bias,
                               const float* __restrict__ g, const float* __restrict__ be,
                               const float* __restrict__ mean, const float* __restrict__ var,
                               float* __restrict__ xo, int N, int HD) {
  int idx = blockIdx.x * blockDim.x + threadIdx.x;
  if (idx >= N * HD) return;
  int f = idx % HD;
  float y = out[idx] + bias[f];
  y = (y - mean[f]) * rsqrtf(var[f] + BN_EPSF) * g[f] + be[f];
  y = (y > 0.f) ? y : expm1f(y);
  xo[idx] = y;
}

// ---------------- classifier: relu(x@cW1+cb1)@cW2+cb2 -> sigmoid -----------
__global__ void k_classifier(const float* __restrict__ x3, const float* __restrict__ cW1,
                             const float* __restrict__ cb1, const float* __restrict__ cW2,
                             const float* __restrict__ cb2, float* __restrict__ out, int N) {
  int n = blockIdx.x * blockDim.x + threadIdx.x;
  if (n >= N) return;
  const float* xr = x3 + (size_t)n * 32;
  float acc = cb2[0];
#pragma unroll
  for (int j = 0; j < 16; ++j) {
    float hsum = cb1[j];
#pragma unroll
    for (int k = 0; k < 32; ++k) hsum += xr[k] * cW1[k * 16 + j];
    hsum = fmaxf(hsum, 0.f);
    acc += hsum * cW2[j];
  }
  out[n] = 1.f / (1.f + expf(-acc));
}

extern "C" void kernel_launch(void* const* d_in, const int* in_sizes, int n_in,
                              void* d_out, int out_size, void* d_ws, size_t ws_size,
                              hipStream_t stream) {
  const float*     x   = (const float*)d_in[0];
  const long long* ei  = (const long long*)d_in[1];
  const float* W1  = (const float*)d_in[2];
  const float* as1 = (const float*)d_in[3];
  const float* ad1 = (const float*)d_in[4];
  const float* b1  = (const float*)d_in[5];
  const float* g1  = (const float*)d_in[6];
  const float* be1 = (const float*)d_in[7];
  const float* m1  = (const float*)d_in[8];
  const float* v1  = (const float*)d_in[9];
  const float* W2  = (const float*)d_in[10];
  const float* as2 = (const float*)d_in[11];
  const float* ad2 = (const float*)d_in[12];
  const float* b2  = (const float*)d_in[13];
  const float* g2  = (const float*)d_in[14];
  const float* be2 = (const float*)d_in[15];
  const float* m2  = (const float*)d_in[16];
  const float* v2  = (const float*)d_in[17];
  const float* W3  = (const float*)d_in[18];
  const float* as3 = (const float*)d_in[19];
  const float* ad3 = (const float*)d_in[20];
  const float* b3  = (const float*)d_in[21];
  const float* g3  = (const float*)d_in[22];
  const float* be3 = (const float*)d_in[23];
  const float* m3  = (const float*)d_in[24];
  const float* v3  = (const float*)d_in[25];
  const float* cW1 = (const float*)d_in[26];
  const float* cb1 = (const float*)d_in[27];
  const float* cW2 = (const float*)d_in[28];
  const float* cb2 = (const float*)d_in[29];

  const int N  = in_sizes[0] / 8;   // IN_DIM = 8
  const int E  = in_sizes[1] / 2;   // edge_index is [2, E]
  const int EP = E + N;             // edges + self-loops

  // ---- workspace carve (256B aligned chunks) ----
  char* p = (char*)d_ws;
  auto take = [&](size_t bytes) -> char* {
    char* r = p;
    p += (bytes + 255) & ~(size_t)255;
    return r;
  };
  float*    hbuf   = (float*)take((size_t)N * 128 * 4);
  float*    outbuf = (float*)take((size_t)N * 128 * 4);
  float*    als    = (float*)take((size_t)N * 4 * 4);
  float*    ald    = (float*)take((size_t)N * 4 * 4);
  float*    mb     = (float*)take((size_t)N * 4 * 4);
  float*    den    = (float*)take((size_t)N * 4 * 4);
  float*    x3     = (float*)take((size_t)N * 32 * 4);
  _Float16* xh     = (_Float16*)take((size_t)N * 128 * 2);
  _Float16* Wt     = (_Float16*)take((size_t)128 * 128 * 2);   // transposed f16 weights

  const int B = 256;                          // 8 waves per workgroup (wave32)
  auto cdiv = [](int a, int b) { return (a + b - 1) / b; };

  // ================= Layer 1: GAT(8 -> 4x32, concat) =================
  k_gemm_in8<<<cdiv(N * 128, B), B, 0, stream>>>(x, W1, hbuf, N, 128);
  k_att_dots<<<cdiv(N * 4, B), B, 0, stream>>>(hbuf, as1, ad1, als, ald, N, 4);
  k_init<<<cdiv(N * 128, B), B, 0, stream>>>(outbuf, mb, den, N, 128, 4);
  k_edge_max<<<cdiv(EP, B), B, 0, stream>>>(ei, E, N, als, ald, mb, 4);
  k_edge_den<<<cdiv(EP, B), B, 0, stream>>>(ei, E, N, als, ald, mb, den, 4);
  k_edge_agg<<<cdiv(EP, 8), B, 0, stream>>>(ei, E, N, als, ald, mb, den, hbuf, outbuf, 4, 128);
  k_finalize_h16<<<cdiv(N * 128, B), B, 0, stream>>>(outbuf, b1, g1, be1, m1, v1, xh, N, 128);

  // ================= Layer 2: GAT(128 -> 4x32, concat), WMMA GEMM =====
  k_transpose_f16<<<cdiv(128 * 128, B), B, 0, stream>>>(W2, Wt, 128, 128);
  int tiles2 = cdiv(N, 16) * (128 / 16);
  k_gemm_wmma<<<cdiv(tiles2, 8), B, 0, stream>>>(xh, Wt, hbuf, N, 128, 128, tiles2);
  k_att_dots<<<cdiv(N * 4, B), B, 0, stream>>>(hbuf, as2, ad2, als, ald, N, 4);
  k_init<<<cdiv(N * 128, B), B, 0, stream>>>(outbuf, mb, den, N, 128, 4);
  k_edge_max<<<cdiv(EP, B), B, 0, stream>>>(ei, E, N, als, ald, mb, 4);
  k_edge_den<<<cdiv(EP, B), B, 0, stream>>>(ei, E, N, als, ald, mb, den, 4);
  k_edge_agg<<<cdiv(EP, 8), B, 0, stream>>>(ei, E, N, als, ald, mb, den, hbuf, outbuf, 4, 128);
  k_finalize_h16<<<cdiv(N * 128, B), B, 0, stream>>>(outbuf, b2, g2, be2, m2, v2, xh, N, 128);

  // ================= Layer 3: GAT(128 -> 32, 1 head), WMMA GEMM =======
  k_transpose_f16<<<cdiv(128 * 32, B), B, 0, stream>>>(W3, Wt, 128, 32);
  int tiles3 = cdiv(N, 16) * (32 / 16);
  k_gemm_wmma<<<cdiv(tiles3, 8), B, 0, stream>>>(xh, Wt, hbuf, N, 32, 128, tiles3);
  k_att_dots<<<cdiv(N * 1, B), B, 0, stream>>>(hbuf, as3, ad3, als, ald, N, 1);
  k_init<<<cdiv(N * 32, B), B, 0, stream>>>(outbuf, mb, den, N, 32, 1);
  k_edge_max<<<cdiv(EP, B), B, 0, stream>>>(ei, E, N, als, ald, mb, 1);
  k_edge_den<<<cdiv(EP, B), B, 0, stream>>>(ei, E, N, als, ald, mb, den, 1);
  k_edge_agg<<<cdiv(EP, 8), B, 0, stream>>>(ei, E, N, als, ald, mb, den, hbuf, outbuf, 1, 32);
  k_finalize_f32<<<cdiv(N * 32, B), B, 0, stream>>>(outbuf, b3, g3, be3, m3, v3, x3, N, 32);

  // ================= Classifier ========================================
  k_classifier<<<cdiv(N, B), B, 0, stream>>>(x3, cW1, cb1, cW2, cb2, (float*)d_out, N);
}
